// TransformerBlock_14989435863669
// MI455X (gfx1250) — compile-verified, matrix operand order checked by reading, not run on hardware
//
#include <hip/hip_runtime.h>
#include <hip/hip_bf16.h>

typedef unsigned short u16;
typedef __attribute__((ext_vector_type(16))) __bf16 v16bf;
typedef __attribute__((ext_vector_type(8)))  float  v8f;
typedef __attribute__((ext_vector_type(4)))  unsigned int v4u;
typedef __attribute__((ext_vector_type(8)))  int v8i;
typedef __attribute__((ext_vector_type(4)))  int v4i;

#define Bb  2
#define Tt  2048
#define Dd  2048
#define Hh  16
#define HDd 128
#define FFf 8192
#define Mm  (Bb*Tt)

__device__ __forceinline__ u16 f2bf(float f) {
  unsigned u = __float_as_uint(f);
  u += 0x7fffu + ((u >> 16) & 1u);           // round-to-nearest-even
  return (u16)(u >> 16);
}
__device__ __forceinline__ float bf2f(u16 h) {
  return __uint_as_float(((unsigned)h) << 16);
}

union FragB { v16bf v; uint4 u[2]; u16 s[16]; };

// ------------------------------------------------------------------ TDM issue
// 2-D tile load via Tensor Data Mover (ISA 08_async_tensor §8 D# layout).
// tensor dims == tile dims (tile start baked into global_addr, no OOB needed).
// data_size = 2 bytes. pad_enable inserts `pad_amount`+1 dwords every
// 2^(pad_interval+1) dwords -> produces the padded LDS stride in-flight.
__device__ __forceinline__ void tdm_load_tile(u16* lds, const u16* gsrc,
                                              unsigned tile0, unsigned tile1,
                                              unsigned long long stride0,
                                              unsigned pad_interval,
                                              unsigned pad_amount) {
  unsigned long long ga = (unsigned long long)(const void*)gsrc;
  unsigned laddr = (unsigned)(unsigned long long)(void*)lds;
  v4u g0;
  g0[0] = 1u;                                             // count=1, user desc
  g0[1] = laddr;                                          // lds_addr
  g0[2] = (unsigned)ga;                                   // global_addr[31:0]
  g0[3] = (unsigned)((ga >> 32) & 0x1ffffffull) | (2u << 30); // [56:32] | type=2
  v8i g1;
  g1[0] = (int)((1u << 16) | (1u << 20) |                 // data_size=2B, pad_en
                (pad_interval << 22) | (pad_amount << 25));
  g1[1] = (int)((tile0 & 0xffffu) << 16);                 // tensor_dim0[15:0]
  g1[2] = (int)((tile0 >> 16) | ((tile1 & 0xffffu) << 16));// dim0 hi | dim1 lo
  g1[3] = (int)((tile1 >> 16) | ((tile0 & 0xffffu) << 16));// dim1 hi | tile_dim0
  g1[4] = (int)(tile1 & 0xffffu);                         // tile_dim1 (tile_dim2=0)
  g1[5] = (int)(unsigned)(stride0 & 0xffffffffull);       // dim0_stride[31:0]
  g1[6] = (int)(unsigned)((stride0 >> 32) & 0xffffull);   // dim0_stride[47:32]
  g1[7] = 0;
  v4i z4 = {0, 0, 0, 0};
  v8i z8 = {0, 0, 0, 0, 0, 0, 0, 0};
  __builtin_amdgcn_tensor_load_to_lds(g0, g1, z4, z4, z8, 0);
}

// ---------------------------------------------------------------- converts
__global__ void k_cvt(const float* __restrict__ in, u16* __restrict__ out, int n8) {
  int i = blockIdx.x * blockDim.x + threadIdx.x;
  int st = gridDim.x * blockDim.x;
  for (; i < n8; i += st) {
    const float4* p = (const float4*)(in + (size_t)i * 8);
    float4 a = p[0], b = p[1];
    union { uint4 u; u16 s[8]; } r;
    r.s[0] = f2bf(a.x); r.s[1] = f2bf(a.y); r.s[2] = f2bf(a.z); r.s[3] = f2bf(a.w);
    r.s[4] = f2bf(b.x); r.s[5] = f2bf(b.y); r.s[6] = f2bf(b.z); r.s[7] = f2bf(b.w);
    *(uint4*)(out + (size_t)i * 8) = r.u;
  }
}

// ---------------------------------------------------------------- layernorm
__global__ __launch_bounds__(256) void k_ln(const float* __restrict__ x,
                                            const float* __restrict__ g,
                                            const float* __restrict__ be,
                                            u16* __restrict__ out) {
  __shared__ float red[256];
  int row = blockIdx.x;
  const float* xr = x + (size_t)row * Dd;
  float v[8];
  float s = 0.f;
#pragma unroll
  for (int i = 0; i < 8; i++) { v[i] = xr[threadIdx.x + i * 256]; s += v[i]; }
  red[threadIdx.x] = s; __syncthreads();
  for (int st = 128; st > 0; st >>= 1) {
    if (threadIdx.x < st) red[threadIdx.x] += red[threadIdx.x + st];
    __syncthreads();
  }
  float mean = red[0] * (1.0f / Dd);
  __syncthreads();
  float s2 = 0.f;
#pragma unroll
  for (int i = 0; i < 8; i++) { float d = v[i] - mean; s2 += d * d; }
  red[threadIdx.x] = s2; __syncthreads();
  for (int st = 128; st > 0; st >>= 1) {
    if (threadIdx.x < st) red[threadIdx.x] += red[threadIdx.x + st];
    __syncthreads();
  }
  float rstd = rsqrtf(red[0] * (1.0f / Dd) + 1e-5f);
  u16* orow = out + (size_t)row * Dd;
#pragma unroll
  for (int i = 0; i < 8; i++) {
    int c = threadIdx.x + i * 256;
    orow[c] = f2bf((v[i] - mean) * rstd * g[c] + be[c]);
  }
}

// ---------------------------------------------------------------- bf16 WMMA GEMM
// C[M,N] = A[M,K] * W[N,K]^T + bias (+resid). Block tile 128x128, K-step 64.
// TDM stages both tiles into double-buffered LDS; wave0 drives the DMA,
// s_wait_tensorcnt 2 overlaps next-tile DMA with current-tile WMMAs.
#define GBM 128
#define GBN 128
#define GBK 64
#define GSTR 72   // 128B row + 16B TDM pad => conflict-free 16B frag loads

__global__ __launch_bounds__(256) void k_gemm(
    const u16* __restrict__ A, const u16* __restrict__ W,
    const float* __restrict__ bias, const float* __restrict__ resid,
    float* __restrict__ outf, u16* __restrict__ outb,
    int Ndim, int Kdim) {
  __shared__ __align__(16) u16 sA[2][GBM * GSTR];
  __shared__ __align__(16) u16 sB[2][GBN * GSTR];
  int tid = threadIdx.x;
  int bm = blockIdx.y * GBM;
  int bn = blockIdx.x * GBN;
  int w = tid >> 5, lane = tid & 31;
  int wm = (w & 3) * 32, wn = (w >> 2) * 64;
  int lr = lane & 15, lh = lane >> 4, kb = lh * 8;

  v8f acc[2][4] = {};
  const u16* Abase = A + (size_t)bm * Kdim;
  const u16* Wbase = W + (size_t)bn * Kdim;
  int nk = Kdim / GBK;

  if (w == 0) {
    tdm_load_tile(&sA[0][0], Abase, GBK, GBM, (unsigned long long)Kdim, 4u, 3u);
    tdm_load_tile(&sB[0][0], Wbase, GBK, GBN, (unsigned long long)Kdim, 4u, 3u);
  }

  int cur = 0;
  for (int kt = 0; kt < nk; ++kt) {
    if (w == 0) {
      if (kt + 1 < nk) {
        tdm_load_tile(&sA[cur ^ 1][0], Abase + (size_t)(kt + 1) * GBK,
                      GBK, GBM, (unsigned long long)Kdim, 4u, 3u);
        tdm_load_tile(&sB[cur ^ 1][0], Wbase + (size_t)(kt + 1) * GBK,
                      GBK, GBN, (unsigned long long)Kdim, 4u, 3u);
        __builtin_amdgcn_s_wait_tensorcnt(2);   // 2 older (current tile) done
      } else {
        __builtin_amdgcn_s_wait_tensorcnt(0);
      }
    }
    __syncthreads();
    const u16* pa = &sA[cur][0];
    const u16* pb = &sB[cur][0];
#pragma unroll
    for (int c = 0; c < 2; ++c) {             // two 32-deep sub-steps
      FragB fa[2], fb[4];
#pragma unroll
      for (int i = 0; i < 2; i++) {
        const u16* p = &pa[(wm + i * 16 + lr) * GSTR + c * 32 + kb];
        fa[i].u[0] = *(const uint4*)p;
        fa[i].u[1] = *(const uint4*)(p + 16);
      }
#pragma unroll
      for (int j = 0; j < 4; j++) {
        const u16* p = &pb[(wn + j * 16 + lr) * GSTR + c * 32 + kb];
        fb[j].u[0] = *(const uint4*)p;
        fb[j].u[1] = *(const uint4*)(p + 16);
      }
#pragma unroll
      for (int i = 0; i < 2; i++)
#pragma unroll
        for (int j = 0; j < 4; j++)
          acc[i][j] = __builtin_amdgcn_wmma_f32_16x16x32_bf16(
              false, fa[i].v, false, fb[j].v, (short)0, acc[i][j], false, false);
    }
    __syncthreads();
    cur ^= 1;
  }

#pragma unroll
  for (int i = 0; i < 2; i++) {
    int gm0 = bm + wm + i * 16 + lh * 8;
#pragma unroll
    for (int j = 0; j < 4; j++) {
      int gn = bn + wn + j * 16 + lr;
      float bv = bias ? bias[gn] : 0.0f;
#pragma unroll
      for (int e = 0; e < 8; e++) {
        size_t idx = (size_t)(gm0 + e) * Ndim + gn;
        float vvv = acc[i][j][e] + bv;
        if (resid) vvv += resid[idx];
        if (outf) outf[idx] = vvv;
        else      outb[idx] = f2bf(vvv);
      }
    }
  }
}

// ---------------------------------------------------------------- RoPE
// Reference quirk: freqs indexed by HEAD (freqs[:H]), broadcast over T.
__global__ void k_rope(u16* __restrict__ q, u16* __restrict__ k,
                       const float* __restrict__ fr) {
  int i = blockIdx.x * blockDim.x + threadIdx.x;  // pair index over M*(D/2)
  if (i >= Mm * (Dd / 2)) return;
  int col2 = i & (Dd / 2 - 1);
  int row = i >> 10;                 // D/2 == 1024
  int h = col2 >> 6;
  int p = col2 & 63;
  float c = fr[(h * 64 + p) * 2 + 0];
  float s = fr[(h * 64 + p) * 2 + 1];
  size_t base = (size_t)row * Dd + h * HDd + 2 * p;
  float a = bf2f(q[base]), b2 = bf2f(q[base + 1]);
  q[base]     = f2bf(a * c - b2 * s);
  q[base + 1] = f2bf(a * s + b2 * c);
  a = bf2f(k[base]); b2 = bf2f(k[base + 1]);
  k[base]     = f2bf(a * c - b2 * s);
  k[base + 1] = f2bf(a * s + b2 * c);
}

// ---------------------------------------------------------------- flash attention
// Block = (b, h, 128 queries). 8 waves, each owns 16 query rows.
// K tile staged by TDM; V staged manually (transposed); P via LDS round-trip.
#define AQ 128
#define ASTR 40
__global__ __launch_bounds__(256) void k_attn(
    const u16* __restrict__ q, const u16* __restrict__ k,
    const u16* __restrict__ v, u16* __restrict__ out) {
  __shared__ __align__(16) u16 sK[32 * 136];        // K tile  [kk][d], TDM-padded
  __shared__ __align__(16) u16 sVt[128 * ASTR];     // V^T     [d][kk]
  __shared__ __align__(16) u16 sP[8 * 16 * ASTR];   // per-wave P tile [row][col]
  int tid = threadIdx.x;
  int w = tid >> 5, lane = tid & 31, lr = lane & 15, lh = lane >> 4, kb = lh * 8;
  int q0 = blockIdx.x * AQ;
  int h  = blockIdx.y;
  int b  = blockIdx.z;
  size_t rowbase = (size_t)b * Tt;
  int headoff = h * HDd;

  FragB fq[4];
  {
    int qr = q0 + w * 16 + lr;
    const u16* qp = q + (rowbase + qr) * Dd + headoff;
#pragma unroll
    for (int c = 0; c < 4; c++) {
      fq[c].u[0] = *(const uint4*)(qp + c * 32 + kb);
      fq[c].u[1] = *(const uint4*)(qp + c * 32 + kb + 16);
    }
  }

  v8f o[8] = {};
  float ms[8], ls[8];
#pragma unroll
  for (int j = 0; j < 8; j++) { ms[j] = -1e30f; ls[j] = 0.f; }

  int srow = tid >> 3, schunk = (tid & 7) * 16;
  const float scale = 0.088388347648318447f;   // 1/sqrt(128)

  for (int kv0 = 0; kv0 < q0 + AQ; kv0 += 32) {
    __syncthreads();
    if (w == 0)   // 32 rows x 128 cols, 256B rows + 16B pad -> stride 136 halves
      tdm_load_tile(&sK[0], k + (rowbase + kv0) * Dd + headoff,
                    128u, 32u, (unsigned long long)Dd, 5u, 3u);
    {
      const u16* vp = v + (rowbase + kv0 + srow) * Dd + headoff + schunk;
      FragB tv; tv.u[0] = *(const uint4*)vp; tv.u[1] = *(const uint4*)(vp + 8);
#pragma unroll
      for (int e = 0; e < 16; e++)
        sVt[(schunk + e) * ASTR + srow] = tv.s[e];
    }
    if (w == 0) __builtin_amdgcn_s_wait_tensorcnt(0);
    __syncthreads();

    // S = Q K^T : two 16x16 tiles covering 32 keys, K-dim = HD (4 x 32)
    v8f sacc[2] = {};
#pragma unroll
    for (int t = 0; t < 2; t++) {
#pragma unroll
      for (int c = 0; c < 4; c++) {
        FragB fk;
        const u16* p = &sK[(t * 16 + lr) * 136 + c * 32 + kb];
        fk.u[0] = *(const uint4*)p;
        fk.u[1] = *(const uint4*)(p + 16);
        sacc[t] = __builtin_amdgcn_wmma_f32_16x16x32_bf16(
            false, fq[c].v, false, fk.v, (short)0, sacc[t], false, false);
      }
    }

    // online softmax (rows live in 16-lane halves)
    int qrow = q0 + w * 16 + lh * 8;
#pragma unroll
    for (int j = 0; j < 8; j++) {
      float s0 = sacc[0][j] * scale;
      float s1 = sacc[1][j] * scale;
      int qi = qrow + j;
      if (kv0 + lr      > qi) s0 = -1e30f;   // causal via select, EXEC stays full
      if (kv0 + 16 + lr > qi) s1 = -1e30f;
      float mx = fmaxf(s0, s1);
      mx = fmaxf(mx, __shfl_xor(mx, 1, 32));
      mx = fmaxf(mx, __shfl_xor(mx, 2, 32));
      mx = fmaxf(mx, __shfl_xor(mx, 4, 32));
      mx = fmaxf(mx, __shfl_xor(mx, 8, 32));
      float mnew = fmaxf(ms[j], mx);
      float corr = __expf(ms[j] - mnew);
      float p0 = __expf(s0 - mnew);
      float p1 = __expf(s1 - mnew);
      float rs = p0 + p1;
      rs += __shfl_xor(rs, 1, 32);
      rs += __shfl_xor(rs, 2, 32);
      rs += __shfl_xor(rs, 4, 32);
      rs += __shfl_xor(rs, 8, 32);
      ls[j] = ls[j] * corr + rs;
      ms[j] = mnew;
#pragma unroll
      for (int c = 0; c < 8; c++) o[c][j] *= corr;
      sP[(w * 16 + lh * 8 + j) * ASTR + lr]      = f2bf(p0);
      sP[(w * 16 + lh * 8 + j) * ASTR + 16 + lr] = f2bf(p1);
    }
    asm volatile("s_wait_dscnt 0" ::: "memory");

    // O += P @ V  (P: 16x32 A-frag from LDS; V^T columns contiguous)
    FragB fp;
    {
      const u16* p = &sP[(w * 16 + lr) * ASTR + kb];
      fp.u[0] = *(const uint4*)p;
      fp.u[1] = *(const uint4*)(p + 16);
    }
#pragma unroll
    for (int c = 0; c < 8; c++) {
      FragB fv;
      const u16* p = &sVt[(c * 16 + lr) * ASTR + kb];
      fv.u[0] = *(const uint4*)p;
      fv.u[1] = *(const uint4*)(p + 16);
      o[c] = __builtin_amdgcn_wmma_f32_16x16x32_bf16(
          false, fp.v, false, fv.v, (short)0, o[c], false, false);
    }
  }

  int qrow = q0 + w * 16 + lh * 8;
#pragma unroll
  for (int c = 0; c < 8; c++) {
    int col = headoff + c * 16 + lr;
#pragma unroll
    for (int j = 0; j < 8; j++)
      out[(rowbase + qrow + j) * Dd + col] = f2bf(o[c][j] / ls[j]);
  }
}

// ---------------------------------------------------------------- silu(g)*u
__global__ void k_silu(const u16* __restrict__ gi, const u16* __restrict__ ui,
                       u16* __restrict__ out, int n) {
  int i = blockIdx.x * blockDim.x + threadIdx.x;
  int st = gridDim.x * blockDim.x;
  for (; i < n; i += st) {
    float g = bf2f(gi[i]);
    float s = g / (1.f + __expf(-g));
    out[i] = f2bf(s * bf2f(ui[i]));
  }
}

// ---------------------------------------------------------------- launch
extern "C" void kernel_launch(void* const* d_in, const int* in_sizes, int n_in,
                              void* d_out, int out_size, void* d_ws, size_t ws_size,
                              hipStream_t stream) {
  const float* x   = (const float*)d_in[0];
  const float* rp  = (const float*)d_in[1];
  const float* Wq  = (const float*)d_in[2];  const float* bq = (const float*)d_in[3];
  const float* Wk  = (const float*)d_in[4];  const float* bk = (const float*)d_in[5];
  const float* Wv  = (const float*)d_in[6];  const float* bv = (const float*)d_in[7];
  const float* Wo  = (const float*)d_in[8];  const float* bo = (const float*)d_in[9];
  const float* g1  = (const float*)d_in[10]; const float* be1 = (const float*)d_in[11];
  const float* g2  = (const float*)d_in[12]; const float* be2 = (const float*)d_in[13];
  const float* Wg  = (const float*)d_in[14]; const float* bg = (const float*)d_in[15];
  const float* Wu  = (const float*)d_in[16]; const float* bu = (const float*)d_in[17];
  const float* Wd  = (const float*)d_in[18]; const float* bd = (const float*)d_in[19];

  char* ws = (char*)d_ws;
  size_t off = 0;
  auto carve = [&](size_t bytes) -> char* {
    char* p = ws + off;
    off += (bytes + 255) & ~(size_t)255;
    return p;
  };
  u16* xn1  = (u16*)carve((size_t)Mm * Dd * 2);
  u16* qb   = (u16*)carve((size_t)Mm * Dd * 2);
  u16* kbuf = (u16*)carve((size_t)Mm * Dd * 2);
  u16* vbuf = (u16*)carve((size_t)Mm * Dd * 2);
  float* x1 = (float*)carve((size_t)Mm * Dd * 4);
  u16* gbuf = (u16*)carve((size_t)Mm * FFf * 2);
  u16* ubuf = (u16*)carve((size_t)Mm * FFf * 2);
  u16* wqb  = (u16*)carve((size_t)Dd * Dd * 2);
  u16* wkb  = (u16*)carve((size_t)Dd * Dd * 2);
  u16* wvb  = (u16*)carve((size_t)Dd * Dd * 2);
  u16* wob  = (u16*)carve((size_t)Dd * Dd * 2);
  u16* wgb  = (u16*)carve((size_t)FFf * Dd * 2);
  u16* wub  = (u16*)carve((size_t)FFf * Dd * 2);
  u16* wdb  = (u16*)carve((size_t)Dd * FFf * 2);

  const int thr = 256;
  k_cvt<<<2048, thr, 0, stream>>>(Wq, wqb, Dd * Dd / 8);
  k_cvt<<<2048, thr, 0, stream>>>(Wk, wkb, Dd * Dd / 8);
  k_cvt<<<2048, thr, 0, stream>>>(Wv, wvb, Dd * Dd / 8);
  k_cvt<<<2048, thr, 0, stream>>>(Wo, wob, Dd * Dd / 8);
  k_cvt<<<4096, thr, 0, stream>>>(Wg, wgb, FFf * Dd / 8);
  k_cvt<<<4096, thr, 0, stream>>>(Wu, wub, FFf * Dd / 8);
  k_cvt<<<4096, thr, 0, stream>>>(Wd, wdb, Dd * FFf / 8);

  k_ln<<<Mm, thr, 0, stream>>>(x, g1, be1, xn1);

  dim3 gD(Dd / GBN, Mm / GBM);     // (16, 32)
  k_gemm<<<gD, thr, 0, stream>>>(xn1, wqb, bq, nullptr, nullptr, qb,   Dd, Dd);
  k_gemm<<<gD, thr, 0, stream>>>(xn1, wkb, bk, nullptr, nullptr, kbuf, Dd, Dd);
  k_gemm<<<gD, thr, 0, stream>>>(xn1, wvb, bv, nullptr, nullptr, vbuf, Dd, Dd);

  k_rope<<<(Mm * (Dd / 2)) / thr, thr, 0, stream>>>(qb, kbuf, rp);

  u16* ao = xn1;  // xn1 dead after V projection
  k_attn<<<dim3(Tt / AQ, Hh, Bb), thr, 0, stream>>>(qb, kbuf, vbuf, ao);

  k_gemm<<<gD, thr, 0, stream>>>(ao, wob, bo, x, x1, nullptr, Dd, Dd);

  u16* xn2 = qb;  // q dead after attention
  k_ln<<<Mm, thr, 0, stream>>>(x1, g2, be2, xn2);

  dim3 gF(FFf / GBN, Mm / GBM);    // (64, 32)
  k_gemm<<<gF, thr, 0, stream>>>(xn2, wgb, bg, nullptr, nullptr, gbuf, FFf, Dd);
  k_gemm<<<gF, thr, 0, stream>>>(xn2, wub, bu, nullptr, nullptr, ubuf, FFf, Dd);

  k_silu<<<4096, thr, 0, stream>>>(gbuf, ubuf, gbuf, Mm * FFf);

  k_gemm<<<gD, thr, 0, stream>>>(gbuf, wdb, bd, x1, (float*)d_out, nullptr, Dd, FFf);
}
